// GroupRouting_90915867722197
// MI455X (gfx1250) — matrix operand order repaired and checked by reading, not compiled, not run on hardware
//
#include <hip/hip_runtime.h>
#include <cstdint>

// ---------------------------------------------------------------------------
// GroupRouting on MI455X (gfx1250, wave32).
//   conv: grouped 3x3, 8 groups, 32 in-ch/group, 32 filters/group, SAME pad.
//   routing collapses to an 8x8 per-batch Gram matrix S => alpha2,
//   out = sum_g alpha2[b,g] * conv_g + bias.
//
// Per conv block (16 positions x all 8 groups, 8 waves):
//   1) GLOBAL_LOAD_ASYNC_TO_LDS_B128 stages the fp32 x tile (3x18x256) into
//      LDS with no VGPR round trip (ASYNCcnt + s_wait_asynccnt).
//   2) one-pass fp32->bf16 conversion into a bank-padded LDS tile
//      (264-elem col stride => conflict-free ds_load_b128 A-fragments).
//   3) 18 x v_wmma_f32_16x16x32_bf16 per wave (9 taps x 2 N-tiles, K=32).
// The 55KB fp32 staging region is dead after step 2 and is aliased by the
// Gram / output-accum buffers.
// ---------------------------------------------------------------------------

typedef __bf16  v16bf __attribute__((ext_vector_type(16)));
typedef __bf16  v8bf  __attribute__((ext_vector_type(8)));
typedef float   v8f   __attribute__((ext_vector_type(8)));

#define HDIM 128
#define WDIM 128
#define CIN  256
#define NGRP 8
#define NFLT 32

#define NCOL 18                     // 16 positions + halo
#define COLS 264                    // padded bf16 col stride (bank-friendly)
#define NCHUNK 3456                 // 3*18*256 fp32 / 4 per 16B chunk
#define STAGE_BYTES 55296           // 3456 * 16
#define XBF_OFF STAGE_BYTES         // bf16 tile after staging region
#define SMEM_BYTES (STAGE_BYTES + 3 * NCOL * COLS * 2)   // 83808

// ------------------------- ws layout ---------------------------------------
//   [0, 4096)      : S          16 * 64 floats
//   [4096, 4608)   : alpha2     16 * 8 floats
//   [8192, ...)    : wfrag bf16 8 groups * 9 taps * 2 ntiles * 512 halfs
// ---------------------------------------------------------------------------

__global__ void zero_S_kernel(float* __restrict__ S) {
    const int t = threadIdx.x;
    S[t] = 0.f; S[t + 256] = 0.f; S[t + 512] = 0.f; S[t + 768] = 0.f;
}

// Pack W[3,3,256,32] fp32 -> bf16 B-fragments, lane-contiguous.
// Fragment fi = (g*9 + tap)*2 + ntile; element (lane, j):
//   N (filter) = ntile*16 + (lane & 15), K (chan in group) = (j<8 ? (lane>>4)*8 + j : 16 + (lane>>4)*8 + j-8)
__global__ void prep_wfrag_kernel(const float* __restrict__ W, __bf16* __restrict__ wf) {
    const int flat = blockIdx.x * 256 + threadIdx.x;   // 73728 total
    const int j    = flat & 15;
    const int lane = (flat >> 4) & 31;
    const int fi   = flat >> 9;
    const int nt   = fi & 1;
    const int gt   = fi >> 1;
    const int tap  = gt % 9;
    const int g    = gt / 9;
    const int f    = nt * 16 + (lane & 15);
    const int kk   = (j < 8) ? (lane >> 4) * 8 + j : 16 + (lane >> 4) * 8 + (j - 8);
    wf[flat] = (__bf16)W[(tap * CIN + g * 32 + kk) * NFLT + f];
}

// Async-stage the fp32 halo tile into LDS, then convert once to padded bf16.
__device__ __forceinline__ void stage_tile(const float* __restrict__ x,
                                           int b, int h, int w0,
                                           char* smem, int tid) {
    const uint32_t lds_base = (uint32_t)(uintptr_t)smem;   // low 32b = LDS offset
    #pragma unroll
    for (int it = 0; it < 14; ++it) {
        const int ci = it * 256 + tid;
        if (ci < NCHUNK) {
            const int row = ci / 1152;                 // 18*64 chunks per row
            const int rem = ci - row * 1152;
            const int col = rem >> 6;
            const int ch4 = rem & 63;
            int rr = h + row - 1;  rr = rr < 0 ? 0 : (rr > 127 ? 127 : rr);
            int cc = w0 + col - 1; cc = cc < 0 ? 0 : (cc > 127 ? 127 : cc);
            const float* gp = x + ((((size_t)b * HDIM + rr) * WDIM + cc) * CIN) + ch4 * 4;
            if (it == 0) __builtin_prefetch(gp + (size_t)WDIM * CIN, 0, 2);
            const uint32_t la = lds_base + (uint32_t)ci * 16u;
            asm volatile("global_load_async_to_lds_b128 %0, %1, off"
                         :: "v"(la), "v"(gp) : "memory");
        }
    }
    asm volatile("s_wait_asynccnt 0x0" ::: "memory");
    __syncthreads();

    const float* sf  = (const float*)smem;
    __bf16*      xbf = (__bf16*)(smem + XBF_OFF);
    #pragma unroll
    for (int it = 0; it < 14; ++it) {
        const int ci = it * 256 + tid;
        if (ci < NCHUNK) {
            const int row = ci / 1152;
            const int rem = ci - row * 1152;
            const int col = rem >> 6;
            const int ch4 = rem & 63;
            const bool ok = ((unsigned)(h + row - 1)  < (unsigned)HDIM) &&
                            ((unsigned)(w0 + col - 1) < (unsigned)WDIM);
            const float4 q = ((const float4*)sf)[ci];
            __bf16* dst = xbf + (row * NCOL + col) * COLS + ch4 * 4;
            if (ok) { dst[0]=(__bf16)q.x; dst[1]=(__bf16)q.y;
                      dst[2]=(__bf16)q.z; dst[3]=(__bf16)q.w; }
            else    { dst[0]=(__bf16)0.f; dst[1]=(__bf16)0.f;
                      dst[2]=(__bf16)0.f; dst[3]=(__bf16)0.f; }
        }
    }
    __syncthreads();
}

// One wave: conv[pos 0..15, group g, f 0..31] = 18 bf16 WMMAs fed from LDS.
__device__ __forceinline__ void conv_tile(const __bf16* __restrict__ xbf,
                                          const __bf16* __restrict__ wf,
                                          int g, int lane, v8f& c0, v8f& c1) {
    const int half = lane >> 4;          // ISA 16-bit A-fragment half
    const int m    = lane & 15;
    const int cb   = g * 32 + half * 8;
    #pragma unroll
    for (int dh = 0; dh < 3; ++dh) {
        #pragma unroll
        for (int dw = 0; dw < 3; ++dw) {
            const __bf16* pa = xbf + (dh * NCOL + m + dw) * COLS + cb;
            const v8bf lo = *(const v8bf*)(pa);        // K = half*8 .. +7
            const v8bf hi = *(const v8bf*)(pa + 16);   // K = 16+half*8 .. +7
            const v16bf a = __builtin_shufflevector(lo, hi,
                0,1,2,3,4,5,6,7, 8,9,10,11,12,13,14,15);
            const int tap = dh * 3 + dw;
            const __bf16* fb = wf + ((size_t)((g * 9 + tap) * 2)) * 512 + lane * 16;
            const v16bf b0 = *(const v16bf*)(fb);
            const v16bf b1 = *(const v16bf*)(fb + 512);
            c0 = __builtin_amdgcn_wmma_f32_16x16x32_bf16(false, a, false, b0,
                                                         (short)0, c0, false, false);
            c1 = __builtin_amdgcn_wmma_f32_16x16x32_bf16(false, a, false, b1,
                                                         (short)0, c1, false, false);
        }
    }
}

// Pass 1: conv + per-batch 8x8 Gram partials into S.
__global__ void __launch_bounds__(256, 2)
conv_gram_kernel(const float* __restrict__ x, const __bf16* __restrict__ wf,
                 float* __restrict__ S) {
    __shared__ __align__(16) char smem[SMEM_BYTES];
    const int tid  = threadIdx.x;
    const int g    = tid >> 5;
    const int lane = tid & 31;
    const int bid  = blockIdx.x;          // 16*128*8 = 16384 blocks
    const int w0   = (bid & 7) << 4;
    const int h    = (bid >> 3) & 127;
    const int b    = bid >> 10;

    stage_tile(x, b, h, w0, smem, tid);

    v8f c0 = {}, c1 = {};
    conv_tile((const __bf16*)(smem + XBF_OFF), wf, g, lane, c0, c1);

    // staging region is dead; alias it for the Gram buffers
    float* lconv = (float*)smem;          // 8 groups * 512 floats
    float* psum  = (float*)(smem + 16384);
    const int nf = lane & 15;
    const int mo = (lane >> 4) << 3;      // C layout: lanes 16-31 -> M=8+r
    float* lg = lconv + g * 512;
    #pragma unroll
    for (int r = 0; r < 8; ++r) {
        lg[(mo + r) * 32 + nf]      = c0[r];
        lg[(mo + r) * 32 + 16 + nf] = c1[r];
    }
    if (tid < 64) psum[tid] = 0.f;
    __syncthreads();

    const int pair = tid >> 2;
    const int sub  = tid & 3;
    const float* p1 = lconv + (pair >> 3) * 512;
    const float* p2 = lconv + (pair & 7) * 512;
    float s = 0.f;
    for (int i = sub; i < 512; i += 4) s += p1[i] * p2[i];
    atomicAdd(&psum[pair], s);            // ds_add_f32
    __syncthreads();
    if (tid < 64) atomicAdd(&S[b * 64 + tid], psum[tid]);
}

// Pass 2: closed-form routing from S -> alpha2[b,g].
__global__ void routing_kernel(const float* __restrict__ S, float* __restrict__ alpha2) {
    const int t = threadIdx.x;            // 128 threads
    const int b = t >> 3;
    const int g = t & 7;
    const float* Sb = S + b * 64;
    float a1[8];
    float beta1_g = 0.f;
    #pragma unroll
    for (int gg = 0; gg < 8; ++gg) {
        float s = 0.f;
        #pragma unroll
        for (int g2 = 0; g2 < 8; ++g2) s += Sb[gg * 8 + g2];
        const float b1 = 0.5f * s;
        a1[gg] = 1.f / (1.f + expf(-b1));
        if (gg == g) beta1_g = b1;
    }
    float beta2 = beta1_g;
    #pragma unroll
    for (int g2 = 0; g2 < 8; ++g2) beta2 += a1[g2] * Sb[g * 8 + g2];
    alpha2[t] = 1.f / (1.f + expf(-beta2));
}

// Pass 3: recompute conv, combine groups with alpha2, add bias, write out.
__global__ void __launch_bounds__(256, 2)
conv_out_kernel(const float* __restrict__ x, const __bf16* __restrict__ wf,
                const float* __restrict__ alpha2, const float* __restrict__ bias,
                float* __restrict__ out) {
    __shared__ __align__(16) char smem[SMEM_BYTES];
    const int tid  = threadIdx.x;
    const int g    = tid >> 5;
    const int lane = tid & 31;
    const int bid  = blockIdx.x;
    const int w0   = (bid & 7) << 4;
    const int h    = (bid >> 3) & 127;
    const int b    = bid >> 10;

    stage_tile(x, b, h, w0, smem, tid);

    // staging region is dead; alias it for the 16x32 output accumulator
    float* accum = (float*)smem;
    accum[tid]       = bias[tid & 31];
    accum[tid + 256] = bias[tid & 31];
    __syncthreads();

    v8f c0 = {}, c1 = {};
    conv_tile((const __bf16*)(smem + XBF_OFF), wf, g, lane, c0, c1);

    const float a2 = alpha2[b * 8 + g];
    const int nf = lane & 15;
    const int mo = (lane >> 4) << 3;
    #pragma unroll
    for (int r = 0; r < 8; ++r) {
        atomicAdd(&accum[(mo + r) * 32 + nf],      a2 * c0[r]);   // ds_add_f32
        atomicAdd(&accum[(mo + r) * 32 + 16 + nf], a2 * c1[r]);
    }
    __syncthreads();

    const size_t obase = (((size_t)b * HDIM + h) * WDIM + w0) * NFLT;
    out[obase + tid]       = accum[tid];
    out[obase + 256 + tid] = accum[tid + 256];
}

extern "C" void kernel_launch(void* const* d_in, const int* in_sizes, int n_in,
                              void* d_out, int out_size, void* d_ws, size_t ws_size,
                              hipStream_t stream) {
    const float* x    = (const float*)d_in[0];   // [16,128,128,256]
    const float* W    = (const float*)d_in[1];   // [3,3,256,32]
    const float* bias = (const float*)d_in[2];   // [32]
    float* out = (float*)d_out;                  // [16,128,128,1,32]

    char*  ws     = (char*)d_ws;
    float* S      = (float*)ws;                  // 1024 floats
    float* alpha2 = (float*)(ws + 4096);         // 128 floats
    __bf16* wf    = (__bf16*)(ws + 8192);        // 73728 bf16 (144 KB)

    const int nblk = 16 * 128 * 8;               // 16384 conv blocks

    zero_S_kernel    <<<1,   256, 0, stream>>>(S);
    prep_wfrag_kernel<<<288, 256, 0, stream>>>(W, wf);
    conv_gram_kernel <<<nblk, 256, 0, stream>>>(x, wf, S);
    routing_kernel   <<<1,   128, 0, stream>>>(S, alpha2);
    conv_out_kernel  <<<nblk, 256, 0, stream>>>(x, wf, alpha2, bias, out);
}
